// ReLoRALinear_69380901700045
// MI455X (gfx1250) — matrix operand order repaired and checked by reading, not compiled, not run on hardware
//
#include <hip/hip_runtime.h>
#include <cstdint>
#include <cstddef>

#define D_IN  4096
#define D_OUT 4096
#define RANK  8
#define SCALE_F 2.0f   // 16.0 / 8.0

typedef __attribute__((ext_vector_type(16))) _Float16 v16h;
typedef __attribute__((ext_vector_type(8)))  _Float16 v8h;
typedef __attribute__((ext_vector_type(8)))  float    v8f;

// ---------------------------------------------------------------------------
// CDNA5 async global->LDS copy (ASYNCcnt-tracked), per CDNA5 ISA
// §15.18.3 (GLOBAL_LOAD_ASYNC_TO_LDS_B128) and CDNA5 ISA.
// ---------------------------------------------------------------------------
__device__ __forceinline__ void async_b128(void* lds, const void* gaddr) {
  uint32_t l = (uint32_t)(uintptr_t)lds;   // low 32 bits of flat LDS ptr
  asm volatile("global_load_async_to_lds_b128 %0, %1, off"
               :: "v"(l), "v"(gaddr) : "memory");
}
__device__ __forceinline__ void wait_async_le12() {
  asm volatile("s_wait_asynccnt 12" ::: "memory");
}
__device__ __forceinline__ void wait_async_0() {
  asm volatile("s_wait_asynccnt 0" ::: "memory");
}
__device__ __forceinline__ v16h cat8(v8h lo, v8h hi) {
  return __builtin_shufflevector(lo, hi, 0,1,2,3,4,5,6,7,8,9,10,11,12,13,14,15);
}

// ---------------------------------------------------------------------------
// K1: fold SRHT (signs + permutation + FWHT) of A and B into A3 (D_IN x RANK)
// and B3 (D_OUT x RANK). One block per (matrix, rank) column; FWHT-4096 in LDS.
// ---------------------------------------------------------------------------
__global__ __launch_bounds__(256) void srht_fold_kernel(
    const float* __restrict__ A, const float* __restrict__ B,
    const float* __restrict__ in_signs, const float* __restrict__ out_signs,
    const int* __restrict__ in_perm, const int* __restrict__ out_perm,
    float* __restrict__ A3, float* __restrict__ B3) {
  __shared__ float buf[D_IN];
  const int tid = threadIdx.x;
  const bool isA = blockIdx.x < RANK;
  const int r = isA ? (int)blockIdx.x : (int)blockIdx.x - RANK;

  for (int j = tid; j < D_IN; j += 256) {
    if (isA) { int p = in_perm[j];  buf[j] = in_signs[p]  * A[r * D_IN + p]; }
    else     { int p = out_perm[j]; buf[j] = out_signs[p] * B[p * RANK + r]; }
  }
  __syncthreads();
  for (int s = 0; s < 12; ++s) {               // log2(4096) butterfly stages
    const int h = 1 << s;
    for (int idx = tid; idx < D_IN / 2; idx += 256) {
      int i = ((idx >> s) << (s + 1)) | (idx & (h - 1));
      float a = buf[i], c = buf[i + h];
      buf[i] = a + c; buf[i + h] = a - c;
    }
    __syncthreads();
  }
  float* dst = isA ? A3 : B3;
  const float norm = 1.0f / 64.0f;             // 1/sqrt(4096), orthonormal FWHT
  for (int j = tid; j < D_IN; j += 256) dst[j * RANK + r] = buf[j] * norm;
}

// ---------------------------------------------------------------------------
// K2a: x (f32) -> xh (f16), 8 elements/thread, vectorized.
// ---------------------------------------------------------------------------
__global__ __launch_bounds__(256) void cvt_x_kernel(
    const float* __restrict__ x, _Float16* __restrict__ xh) {
  const size_t id = (size_t)blockIdx.x * 256 + threadIdx.x;   // one per 8 elems
  const float4* src = (const float4*)x;
  float4 f0 = src[id * 2 + 0];
  float4 f1 = src[id * 2 + 1];
  v8h h;
  h[0] = (_Float16)f0.x; h[1] = (_Float16)f0.y;
  h[2] = (_Float16)f0.z; h[3] = (_Float16)f0.w;
  h[4] = (_Float16)f1.x; h[5] = (_Float16)f1.y;
  h[6] = (_Float16)f1.z; h[7] = (_Float16)f1.w;
  ((v8h*)xh)[id] = h;
}

// ---------------------------------------------------------------------------
// K2b: W_eff = f16( W + SCALE * B3 @ A3^T ), 8 elements (one o, 8 i's)/thread.
// ---------------------------------------------------------------------------
__global__ __launch_bounds__(256) void build_weff_kernel(
    const float* __restrict__ W, const float* __restrict__ A3,
    const float* __restrict__ B3, _Float16* __restrict__ Weff) {
  const size_t id = (size_t)blockIdx.x * 256 + threadIdx.x;
  const int o  = (int)(id >> 9);          // D_IN/8 = 512 groups per row
  const int i0 = ((int)id & 511) * 8;
  float b3[RANK];
#pragma unroll
  for (int r = 0; r < RANK; ++r) b3[r] = B3[(size_t)o * RANK + r];
  const float* wrow = W + (size_t)o * D_IN + i0;
  v8h h;
#pragma unroll
  for (int e = 0; e < 8; ++e) {
    const float* a3 = A3 + (size_t)(i0 + e) * RANK;
    float dot = 0.f;
#pragma unroll
    for (int r = 0; r < RANK; ++r) dot += a3[r] * b3[r];
    h[e] = (_Float16)(wrow[e] + SCALE_F * dot);
  }
  ((v8h*)Weff)[id] = h;
}

// ---------------------------------------------------------------------------
// K3: out[m,n] = sum_k xh[m,k]*Weff[n,k] + bias[n]  via v_wmma_f32_16x16x32_f16.
// Block tile 256x128, 8 waves (4x2), wave tile 64x64 (4x4 WMMA tiles -> one
// ds_load_b128 per WMMA), K-step 64, double-buffered LDS fed by async copies.
// ---------------------------------------------------------------------------
#define TM 256
#define TN 128
#define TK 64
#define LDT 72   // padded row length (halfs): 144B stride -> conflict-free frags

__global__ __launch_bounds__(256) void gemm_wmma_kernel(
    const _Float16* __restrict__ xh, const _Float16* __restrict__ wh,
    const float* __restrict__ bias, float* __restrict__ out) {
  __shared__ _Float16 Ash[2][TM][LDT];   // 73,728 B
  __shared__ _Float16 Bsh[2][TN][LDT];   // 36,864 B

  const int tid  = threadIdx.x;
  const int lane = tid & 31;
  const int wave = tid >> 5;
  const int wm = wave >> 1;                 // 0..3 : 64-row slab
  const int wn = wave & 1;                  // 0..1 : 64-col slab
  const int m0 = blockIdx.y * TM;
  const int n0 = blockIdx.x * TN;

  // ISA §7.12.2 fragment lane mappings (16-bit, wave32):
  const int lrow  = lane & 15;
  const int asel  = (lane & 16) ? 8  : 0;   // A: K = asel + {0..7, 16..23}
  const int bsel  = (lane & 16) ? 8  : 0;   // B: K = bsel + {0..7, 16..23}
  const int mhalf = (lane & 16) ? 8  : 0;   // D: M = r + mhalf

  v8f acc[4][4];
#pragma unroll
  for (int a = 0; a < 4; ++a)
#pragma unroll
    for (int b = 0; b < 4; ++b)
      acc[a][b] = (v8f){0.f,0.f,0.f,0.f,0.f,0.f,0.f,0.f};

  // 12 async b128 per thread per stage: 8 for the A tile + 4 for the B tile.
  auto issue_tile = [&](int buf, int kk) {
    const _Float16* sA = xh + (size_t)m0 * D_IN + (size_t)kk * TK;
    const _Float16* sB = wh + (size_t)n0 * D_IN + (size_t)kk * TK;
#pragma unroll
    for (int t = 0; t < 8; ++t) {           // 2048 16B-chunks: A tile
      const int c   = tid + t * 256;
      const int row = c >> 3;
      const int col = (c & 7) * 8;
      async_b128(&Ash[buf][row][col], sA + (size_t)row * D_IN + col);
    }
#pragma unroll
    for (int t = 0; t < 4; ++t) {           // 1024 16B-chunks: B tile
      const int c   = tid + t * 256;
      const int row = c >> 3;
      const int col = (c & 7) * 8;
      async_b128(&Bsh[buf][row][col], sB + (size_t)row * D_IN + col);
    }
  };

  issue_tile(0, 0);

  const int nK = D_IN / TK;                 // 64 K-stages
  for (int kk = 0; kk < nK; ++kk) {
    const int cur = kk & 1;
    if (kk + 1 < nK) {
      issue_tile(cur ^ 1, kk + 1);          // prefetch next stage
      wait_async_le12();                    // current stage's 12 done (in-order)
    } else {
      wait_async_0();
    }
    __syncthreads();

#pragma unroll
    for (int ks = 0; ks < 2; ++ks) {        // two K=32 WMMA steps
      const int k0 = ks * 32;
      v16h af[4], bf[4];
#pragma unroll
      for (int a = 0; a < 4; ++a) {
        const _Float16* base = &Ash[cur][wm * 64 + a * 16 + lrow][k0 + asel];
        v8h lo = *(const v8h*)(base);       // K = asel + 0..7
        v8h hi = *(const v8h*)(base + 16);  // K = asel + 16..23
        af[a] = cat8(lo, hi);
      }
#pragma unroll
      for (int b = 0; b < 4; ++b) {
        const _Float16* base = &Bsh[cur][wn * 64 + b * 16 + lrow][k0 + bsel];
        v8h lo = *(const v8h*)(base);       // K = bsel + 0..7
        v8h hi = *(const v8h*)(base + 16);  // K = bsel + 16..23
        bf[b] = cat8(lo, hi);
      }
#pragma unroll
      for (int a = 0; a < 4; ++a)
#pragma unroll
        for (int b = 0; b < 4; ++b)
          acc[a][b] = __builtin_amdgcn_wmma_f32_16x16x32_f16(
              false, af[a], false, bf[b], (short)0, acc[a][b], false, false);
    }
    __syncthreads();                        // tile fully consumed before reuse
  }

  // Epilogue: D layout -> global, fused bias.
  float biasv[4];
  int   ncol[4];
#pragma unroll
  for (int b = 0; b < 4; ++b) {
    ncol[b]  = n0 + wn * 64 + b * 16 + lrow;
    biasv[b] = bias[ncol[b]];
  }
#pragma unroll
  for (int a = 0; a < 4; ++a) {
    const int mbase = m0 + wm * 64 + a * 16 + mhalf;
#pragma unroll
    for (int r = 0; r < 8; ++r) {
      const size_t m = (size_t)(mbase + r);
#pragma unroll
      for (int b = 0; b < 4; ++b)
        out[m * D_OUT + ncol[b]] = acc[a][b][r] + biasv[b];
    }
  }
}

// ---------------------------------------------------------------------------
// Inputs (setup_inputs order): 0 x, 1 W, 2 b, 3 A, 4 B, 5 in_signs,
// 6 out_signs, 7 in_perm, 8 in_inv_perm, 9 out_perm, 10 out_inv_perm.
// Workspace layout: [Weff f16 32MB][xh f16 64MB][A3 128KB][B3 128KB]
// ---------------------------------------------------------------------------
extern "C" void kernel_launch(void* const* d_in, const int* in_sizes, int n_in,
                              void* d_out, int out_size, void* d_ws, size_t ws_size,
                              hipStream_t stream) {
  const float* x         = (const float*)d_in[0];
  const float* W         = (const float*)d_in[1];
  const float* b         = (const float*)d_in[2];
  const float* A         = (const float*)d_in[3];
  const float* B         = (const float*)d_in[4];
  const float* in_signs  = (const float*)d_in[5];
  const float* out_signs = (const float*)d_in[6];
  const int*   in_perm   = (const int*)d_in[7];
  const int*   out_perm  = (const int*)d_in[9];
  float* out = (float*)d_out;

  const int M = in_sizes[0] / D_IN;                       // 8192 tokens

  char* ws = (char*)d_ws;
  _Float16* wh = (_Float16*)ws;                                       // 32 MB
  _Float16* xh = (_Float16*)(ws + (size_t)D_OUT * D_IN * 2);          // 64 MB
  float* A3 = (float*)(ws + (size_t)D_OUT * D_IN * 2 + (size_t)M * D_IN * 2);
  float* B3 = A3 + (size_t)D_IN * RANK;

  srht_fold_kernel<<<2 * RANK, 256, 0, stream>>>(
      A, B, in_signs, out_signs, in_perm, out_perm, A3, B3);

  const unsigned gx = (unsigned)(((size_t)M * D_IN / 8) / 256);       // 16384
  cvt_x_kernel<<<gx, 256, 0, stream>>>(x, xh);

  const unsigned gw = (unsigned)(((size_t)D_OUT * D_IN / 8) / 256);   // 8192
  build_weff_kernel<<<gw, 256, 0, stream>>>(W, A3, B3, wh);

  gemm_wmma_kernel<<<dim3(D_OUT / TN, M / TM), 256, 0, stream>>>(xh, wh, b, out);
}
